// MultiHeadedAttentionWithRelations_42485816492442
// MI455X (gfx1250) — matrix operand order repaired and checked by reading, not compiled, not run on hardware
//
#include <hip/hip_runtime.h>
#include <hip/hip_bf16.h>

typedef __bf16 bf16_t;
typedef __attribute__((ext_vector_type(16))) __bf16 v16bf;
typedef __attribute__((ext_vector_type(8)))  float  v8f;

#define B_  8
#define N_  256
#define D_  512
#define H_  8
#define DK_ 64

// ---------------------------------------------------------------------------
// bf16 bit helpers (unpack two bf16 from a dword)
// ---------------------------------------------------------------------------
__device__ inline float bflo(unsigned u) {
  union { unsigned i; float f; } c; c.i = u << 16; return c.f;
}
__device__ inline float bfhi(unsigned u) {
  union { unsigned i; float f; } c; c.i = u & 0xffff0000u; return c.f;
}

// 64-element bf16 dot product via 8x B128 loads per operand.
__device__ inline float dot64_bf16(const bf16_t* a, const bf16_t* b) {
  const uint4* pa = (const uint4*)a;
  const uint4* pb = (const uint4*)b;
  float s = 0.f;
#pragma unroll
  for (int c = 0; c < 8; ++c) {
    uint4 ua = pa[c], ub = pb[c];
    s = fmaf(bflo(ua.x), bflo(ub.x), s); s = fmaf(bfhi(ua.x), bfhi(ub.x), s);
    s = fmaf(bflo(ua.y), bflo(ub.y), s); s = fmaf(bfhi(ua.y), bfhi(ub.y), s);
    s = fmaf(bflo(ua.z), bflo(ub.z), s); s = fmaf(bfhi(ua.z), bfhi(ub.z), s);
    s = fmaf(bflo(ua.w), bflo(ub.w), s); s = fmaf(bfhi(ua.w), bfhi(ub.w), s);
  }
  return s;
}

// acc[0..31] += p * src[0..31]  (bf16 source, 4x B128 loads)
__device__ inline void axpy32_bf16(float p, const bf16_t* src, float* acc) {
  const uint4* ps = (const uint4*)src;
#pragma unroll
  for (int c = 0; c < 4; ++c) {
    uint4 u = ps[c];
    acc[c * 8 + 0] = fmaf(p, bflo(u.x), acc[c * 8 + 0]);
    acc[c * 8 + 1] = fmaf(p, bfhi(u.x), acc[c * 8 + 1]);
    acc[c * 8 + 2] = fmaf(p, bflo(u.y), acc[c * 8 + 2]);
    acc[c * 8 + 3] = fmaf(p, bfhi(u.y), acc[c * 8 + 3]);
    acc[c * 8 + 4] = fmaf(p, bflo(u.z), acc[c * 8 + 4]);
    acc[c * 8 + 5] = fmaf(p, bfhi(u.z), acc[c * 8 + 5]);
    acc[c * 8 + 6] = fmaf(p, bflo(u.w), acc[c * 8 + 6]);
    acc[c * 8 + 7] = fmaf(p, bfhi(u.w), acc[c * 8 + 7]);
  }
}

// ---------------------------------------------------------------------------
// WMMA operand loaders (CDNA5 16x16x32 bf16 layouts, ISA 7.12.2):
//   A (16x32, 16-bit): lane m=lane&15, g=lane>>4; K runs [g*8,+8) and [16+g*8,+8)
//   B (32x16, 16-bit): lane n=lane&15, g=lane>>4; K runs [g*8,+8) and [16+g*8,+8)
//   C/D (16x16 f32):   VGPR v -> row (v + 8*g), col = lane&15
// ---------------------------------------------------------------------------
__device__ inline v16bf load_a_bf16(const bf16_t* base, int ld, int k0, int lane) {
  int m = lane & 15, g = lane >> 4;
  const bf16_t* p = base + (size_t)m * ld + k0 + g * 8;
  v16bf a;
#pragma unroll
  for (int e = 0; e < 8; ++e) a[e] = p[e];
#pragma unroll
  for (int e = 0; e < 8; ++e) a[8 + e] = p[16 + e];
  return a;
}
__device__ inline v16bf load_a_f32(const float* base, int ld, int k0, int lane) {
  int m = lane & 15, g = lane >> 4;
  const float* p = base + (size_t)m * ld + k0 + g * 8;
  v16bf a;
#pragma unroll
  for (int e = 0; e < 8; ++e) a[e] = (bf16_t)p[e];
#pragma unroll
  for (int e = 0; e < 8; ++e) a[8 + e] = (bf16_t)p[16 + e];
  return a;
}
__device__ inline v16bf load_b_bf16(const bf16_t* base, int ld, int k0, int lane) {
  int n = lane & 15, g = lane >> 4;
  const bf16_t* p = base + (size_t)n * ld + k0 + g * 8;
  v16bf b;
#pragma unroll
  for (int e = 0; e < 8; ++e) b[e] = p[e];
#pragma unroll
  for (int e = 0; e < 8; ++e) b[8 + e] = p[16 + e];
  return b;
}
__device__ inline v16bf load_b_f32(const float* base, int ld, int k0, int lane) {
  int n = lane & 15, g = lane >> 4;
  const float* p = base + (size_t)n * ld + k0 + g * 8;
  v16bf b;
#pragma unroll
  for (int e = 0; e < 8; ++e) b[e] = (bf16_t)p[e];
#pragma unroll
  for (int e = 0; e < 8; ++e) b[8 + e] = (bf16_t)p[16 + e];
  return b;
}
__device__ inline v8f wmma_bf16(v16bf a, v16bf b, v8f c) {
  return __builtin_amdgcn_wmma_f32_16x16x32_bf16(false, a, false, b, (short)0, c,
                                                 false, false);
}

// ---------------------------------------------------------------------------
// K1: q/k/v projections.  [2048 x 512] @ W^T + bias -> bf16 [B,H,N,DK]
// One wave = one 16x16 output tile.  3 matrices * 128 * 32 tiles = 12288.
// ---------------------------------------------------------------------------
__global__ void __launch_bounds__(256) qkv_proj_kernel(
    const float* __restrict__ q_in, const float* __restrict__ k_in,
    const float* __restrict__ v_in, const float* __restrict__ Wq,
    const float* __restrict__ bq, const float* __restrict__ Wk,
    const float* __restrict__ bk, const float* __restrict__ Wv,
    const float* __restrict__ bv, bf16_t* __restrict__ q_o,
    bf16_t* __restrict__ k_o, bf16_t* __restrict__ v_o) {
  int lane = threadIdx.x & 31;
  int tid  = blockIdx.x * 8 + (threadIdx.x >> 5);
  int mat  = tid >> 12;  // 4096 tiles per matrix
  int t    = tid & 4095;
  int mt   = t >> 5;     // 128 row tiles
  int nt   = t & 31;     // 32 col tiles
  const float* src = (mat == 0) ? q_in : (mat == 1) ? k_in : v_in;
  const float* W   = (mat == 0) ? Wq : (mat == 1) ? Wk : Wv;
  const float* bia = (mat == 0) ? bq : (mat == 1) ? bk : bv;
  bf16_t* dst      = (mat == 0) ? q_o : (mat == 1) ? k_o : v_o;

  v8f acc = {0, 0, 0, 0, 0, 0, 0, 0};
  const float* abase = src + (size_t)mt * 16 * D_;
  const float* bbase = W + (size_t)nt * 16 * D_;
  for (int kk = 0; kk < D_; kk += 32) {
    v16bf a = load_a_f32(abase, D_, kk, lane);
    v16bf b = load_b_f32(bbase, D_, kk, lane);
    acc = wmma_bf16(a, b, acc);
  }
  int   ncol = nt * 16 + (lane & 15);
  float bias = bia[ncol];
  int   h = ncol >> 6, dk = ncol & 63;
  int   g = lane >> 4;
#pragma unroll
  for (int v = 0; v < 8; ++v) {
    int m  = mt * 16 + v + 8 * g;
    int bb = m >> 8, n = m & 255;
    dst[((size_t)((bb * H_ + h) * N_ + n)) * DK_ + dk] = (bf16_t)(acc[v] + bias);
  }
}

// ---------------------------------------------------------------------------
// K2: relation projections  [B*N*N,128] @ W^T + bias, ReLU -> bf16 [.,64]
// One wave owns a full 16-row stripe: A-tile loaded ONCE, reused across all
// 4 output column tiles and both weight matrices (rel_k + rel_v fused).
// For st: ov == nullptr -> second WMMA chain skipped (uniform branch).
// ---------------------------------------------------------------------------
__global__ void __launch_bounds__(256) rel_proj_kernel(
    const float* __restrict__ mask, const float* __restrict__ Wk_,
    const float* __restrict__ bk_, const float* __restrict__ Wv_,
    const float* __restrict__ bv_, bf16_t* __restrict__ ok,
    bf16_t* __restrict__ ov) {
  int lane = threadIdx.x & 31;
  int mt   = blockIdx.x * 8 + (threadIdx.x >> 5);  // 32768 row stripes
  v8f accK[4] = {{0, 0, 0, 0, 0, 0, 0, 0}, {0, 0, 0, 0, 0, 0, 0, 0},
                 {0, 0, 0, 0, 0, 0, 0, 0}, {0, 0, 0, 0, 0, 0, 0, 0}};
  v8f accV[4] = {{0, 0, 0, 0, 0, 0, 0, 0}, {0, 0, 0, 0, 0, 0, 0, 0},
                 {0, 0, 0, 0, 0, 0, 0, 0}, {0, 0, 0, 0, 0, 0, 0, 0}};
  const float* abase = mask + (size_t)mt * 16 * 128;
  // warm L2 one block (8 stripes) ahead; speculative, bounds-guarded
  if (mt + 8 < 32768) __builtin_prefetch(abase + (size_t)8 * 16 * 128, 0, 1);
#pragma unroll
  for (int kk = 0; kk < 128; kk += 32) {
    v16bf a = load_a_f32(abase, 128, kk, lane);
#pragma unroll
    for (int nt = 0; nt < 4; ++nt) {
      v16bf b1 = load_b_f32(Wk_ + (size_t)nt * 16 * 128, 128, kk, lane);
      accK[nt] = wmma_bf16(a, b1, accK[nt]);
      if (ov) {
        v16bf b2 = load_b_f32(Wv_ + (size_t)nt * 16 * 128, 128, kk, lane);
        accV[nt] = wmma_bf16(a, b2, accV[nt]);
      }
    }
  }
  int g = lane >> 4;
#pragma unroll
  for (int nt = 0; nt < 4; ++nt) {
    int   c   = nt * 16 + (lane & 15);
    float biK = bk_[c];
#pragma unroll
    for (int v = 0; v < 8; ++v) {
      size_t r = (size_t)mt * 16 + v + 8 * g;
      ok[r * 64 + c] = (bf16_t)fmaxf(accK[nt][v] + biK, 0.f);
    }
    if (ov) {
      float biV = bv_[c];
#pragma unroll
      for (int v = 0; v < 8; ++v) {
        size_t r = (size_t)mt * 16 + v + 8 * g;
        ov[r * 64 + c] = (bf16_t)fmaxf(accV[nt][v] + biV, 0.f);
      }
    }
  }
}

// ---------------------------------------------------------------------------
// K3: box relational embedding (sin/cos) + 64->64 linear + ReLU -> bx bf16.
// ---------------------------------------------------------------------------
__global__ void __launch_bounds__(256) box_embed_kernel(
    const float* __restrict__ boxes, const float* __restrict__ Wbx,
    const float* __restrict__ bbx, bf16_t* __restrict__ bxo) {
  int lane = threadIdx.x & 31;
  int wid  = blockIdx.x * 8 + (threadIdx.x >> 5);  // 32768
  int b    = wid >> 12;
  int rem  = wid & 4095;
  int i    = rem >> 4;
  int jt   = rem & 15;
  int j    = jt * 16 + (lane & 15);

  const float* bi = boxes + (size_t)(b * N_ + i) * 4;
  const float* bj = boxes + (size_t)(b * N_ + j) * 4;
  float cxi = (bi[0] + bi[2]) * 0.5f, cyi = (bi[1] + bi[3]) * 0.5f;
  float wi = bi[2] - bi[0] + 1.f, hi = bi[3] - bi[1] + 1.f;
  float cxj = (bj[0] + bj[2]) * 0.5f, cyj = (bj[1] + bj[3]) * 0.5f;
  float wj = bj[2] - bj[0] + 1.f, hj = bj[3] - bj[1] + 1.f;
  float pos[4];
  pos[0] = __logf(fmaxf(fabsf(cxi - cxj) / wi, 1e-3f));
  pos[1] = __logf(fmaxf(fabsf(cyi - cyj) / hi, 1e-3f));
  pos[2] = __logf(wi / wj);
  pos[3] = __logf(hi / hj);

  float emb[64];
#pragma unroll
  for (int p = 0; p < 4; ++p) {
#pragma unroll
    for (int t = 0; t < 8; ++t) {
      // 1/dim_mat[t] = 1000^(-t/8)
      float invdm = __expf(-(float)t * 0.125f * 6.9077552790f);
      float mul   = pos[p] * 100.f * invdm;
      emb[p * 16 + t]     = __sinf(mul);
      emb[p * 16 + 8 + t] = __cosf(mul);
    }
  }

  int g = lane >> 4, kb = g * 8;
  v8f acc[4] = {{0, 0, 0, 0, 0, 0, 0, 0}, {0, 0, 0, 0, 0, 0, 0, 0},
                {0, 0, 0, 0, 0, 0, 0, 0}, {0, 0, 0, 0, 0, 0, 0, 0}};
  for (int K0 = 0; K0 < 64; K0 += 32) {
    v16bf a;
#pragma unroll
    for (int e = 0; e < 8; ++e) a[e] = (bf16_t)emb[K0 + kb + e];
#pragma unroll
    for (int e = 0; e < 8; ++e) a[8 + e] = (bf16_t)emb[K0 + 16 + kb + e];
#pragma unroll
    for (int nt = 0; nt < 4; ++nt) {
      v16bf bm = load_b_f32(Wbx + (size_t)nt * 16 * 64, 64, K0, lane);
      acc[nt]  = wmma_bf16(a, bm, acc[nt]);
    }
  }
#pragma unroll
  for (int nt = 0; nt < 4; ++nt) {
    int   d    = nt * 16 + (lane & 15);
    float bias = bbx[d];
#pragma unroll
    for (int v = 0; v < 8; ++v) {
      int    jl  = v + 8 * g;
      size_t idx = (((size_t)(b * N_ + i) * N_) + jt * 16 + jl) * DK_ + d;
      bxo[idx]   = (bf16_t)fmaxf(acc[nt][v] + bias, 0.f);
    }
  }
}

// ---------------------------------------------------------------------------
// K4: fused attention.  One workgroup per (b, i-tile); wave h = head h.
// Scores = WMMA(q k^T) + 4 B128-vectorized relation dot-terms; flash softmax
// via LDS score-tile transpose; sigmoid-gated mix with __shfl_xor reduction.
// ---------------------------------------------------------------------------
__global__ void __launch_bounds__(256) attn_kernel(
    const bf16_t* __restrict__ qb, const bf16_t* __restrict__ kb,
    const bf16_t* __restrict__ vb, const bf16_t* __restrict__ relk,
    const bf16_t* __restrict__ relv, const bf16_t* __restrict__ stb,
    const bf16_t* __restrict__ bxb, const float* __restrict__ Wvw,
    const float* __restrict__ bvw, bf16_t* __restrict__ xo) {
  __shared__ float scbuf[8][16][17];
  int lane  = threadIdx.x & 31;
  int h     = threadIdx.x >> 5;
  int b     = blockIdx.x >> 4;
  int itile = blockIdx.x & 15;

  const bf16_t* qhead = qb + (size_t)((b * H_ + h) * N_) * DK_;
  const bf16_t* khead = kb + (size_t)((b * H_ + h) * N_) * DK_;
  const bf16_t* vhead = vb + (size_t)((b * H_ + h) * N_) * DK_;

  v16bf aQ0 = load_a_bf16(qhead + (size_t)itile * 16 * DK_, DK_, 0, lane);
  v16bf aQ1 = load_a_bf16(qhead + (size_t)itile * 16 * DK_, DK_, 32, lane);

  int g    = lane >> 4;
  int jl   = lane & 15;
  int irow = lane >> 1;        // output row this lane owns
  int dh   = (lane & 1) * 32;  // output dim half (64B-aligned offset)
  int igr  = itile * 16 + irow;

  float accv[32], accr[32], accb[32];
#pragma unroll
  for (int d = 0; d < 32; ++d) { accv[d] = 0.f; accr[d] = 0.f; accb[d] = 0.f; }
  float mrun = -1e30f, lrun = 0.f;

  for (int jt = 0; jt < 16; ++jt) {
    v8f sc = {0, 0, 0, 0, 0, 0, 0, 0};
    sc = wmma_bf16(aQ0, load_b_bf16(khead + (size_t)jt * 16 * DK_, DK_, 0, lane), sc);
    sc = wmma_bf16(aQ1, load_b_bf16(khead + (size_t)jt * 16 * DK_, DK_, 32, lane), sc);

    int           jg  = jt * 16 + jl;
    const bf16_t* q_j = qhead + (size_t)jg * DK_;
#pragma unroll
    for (int v = 0; v < 8; ++v) {
      int           ig  = itile * 16 + v + 8 * g;
      const bf16_t* rji = relk + (((size_t)(b * N_ + jg)) * N_ + ig) * DK_;
      const bf16_t* rij = relk + (((size_t)(b * N_ + ig)) * N_ + jg) * DK_;
      const bf16_t* bxp = bxb + (((size_t)(b * N_ + ig)) * N_ + jg) * DK_;
      const bf16_t* stp = stb + (((size_t)(b * N_ + ig)) * N_ + jg) * DK_;
      float t = dot64_bf16(q_j, rji)                          // term2 (transposed)
              + dot64_bf16(khead + (size_t)ig * DK_, rij)     // term3
              + dot64_bf16(qhead + (size_t)ig * DK_, bxp)     // term4 (geometry)
              + dot64_bf16(qhead + (size_t)ig * DK_, stp);    // term5 (structure)
      sc[v] = (sc[v] + t) * 0.125f;  // 1/sqrt(64)
    }

#pragma unroll
    for (int v = 0; v < 8; ++v) scbuf[h][v + 8 * g][jl] = sc[v];
    __syncthreads();

    float rmax = -1e30f;
#pragma unroll
    for (int j = 0; j < 16; ++j) rmax = fmaxf(rmax, scbuf[h][irow][j]);
    float nm   = fmaxf(mrun, rmax);
    float corr = __expf(mrun - nm);
    lrun *= corr;
#pragma unroll
    for (int d = 0; d < 32; ++d) { accv[d] *= corr; accr[d] *= corr; accb[d] *= corr; }

    for (int j = 0; j < 16; ++j) {
      float p = __expf(scbuf[h][irow][j] - nm);
      lrun += p;
      int jg2 = jt * 16 + j;
      axpy32_bf16(p, vhead + (size_t)jg2 * DK_ + dh, accv);
      axpy32_bf16(p, relv + (((size_t)(b * N_ + igr)) * N_ + jg2) * DK_ + dh, accr);
      axpy32_bf16(p, bxb + (((size_t)(b * N_ + igr)) * N_ + jg2) * DK_ + dh, accb);
    }
    mrun = nm;
    __syncthreads();
  }

  float inv = 1.f / lrun;
#pragma unroll
  for (int d = 0; d < 32; ++d) { accv[d] *= inv; accr[d] *= inv; accb[d] *= inv; }

  // gate = sigmoid(Wvw @ [wv;wr;wb] + bvw); partner lane holds other dim-half
  float g0p = 0.f, g1p = 0.f;
#pragma unroll
  for (int d = 0; d < 32; ++d) {
    g0p += Wvw[dh + d] * accv[d] + Wvw[64 + dh + d] * accr[d] +
           Wvw[128 + dh + d] * accb[d];
    g1p += Wvw[192 + dh + d] * accv[d] + Wvw[192 + 64 + dh + d] * accr[d] +
           Wvw[192 + 128 + dh + d] * accb[d];
  }
  g0p += __shfl_xor(g0p, 1, 32);
  g1p += __shfl_xor(g1p, 1, 32);
  float gate0 = 1.f / (1.f + __expf(-(g0p + bvw[0])));
  float gate1 = 1.f / (1.f + __expf(-(g1p + bvw[1])));

  size_t obase = ((size_t)(b * N_ + igr)) * D_ + h * DK_ + dh;
#pragma unroll
  for (int d = 0; d < 32; ++d) {
    xo[obase + d] = (bf16_t)(accv[d] + gate0 * accr[d] + gate1 * accb[d]);
  }
}

// ---------------------------------------------------------------------------
// K5: output projection  x[2048x512](bf16) @ Wo^T + bo -> fp32 out
// ---------------------------------------------------------------------------
__global__ void __launch_bounds__(256) out_proj_kernel(
    const bf16_t* __restrict__ x, const float* __restrict__ Wo,
    const float* __restrict__ bo, float* __restrict__ out) {
  int lane = threadIdx.x & 31;
  int tid  = blockIdx.x * 8 + (threadIdx.x >> 5);  // 4096 tiles
  int mt   = tid >> 5;
  int nt   = tid & 31;
  v8f acc  = {0, 0, 0, 0, 0, 0, 0, 0};
  for (int kk = 0; kk < D_; kk += 32) {
    v16bf a = load_a_bf16(x + (size_t)mt * 16 * D_, D_, kk, lane);
    v16bf b = load_b_f32(Wo + (size_t)nt * 16 * D_, D_, kk, lane);
    acc = wmma_bf16(a, b, acc);
  }
  int   c    = nt * 16 + (lane & 15);
  float bias = bo[c];
  int   g    = lane >> 4;
#pragma unroll
  for (int v = 0; v < 8; ++v) {
    int m = mt * 16 + v + 8 * g;
    out[(size_t)m * D_ + c] = acc[v] + bias;
  }
}

// ---------------------------------------------------------------------------
extern "C" void kernel_launch(void* const* d_in, const int* in_sizes, int n_in,
                              void* d_out, int out_size, void* d_ws,
                              size_t ws_size, hipStream_t stream) {
  const float* query = (const float*)d_in[0];
  const float* key   = (const float*)d_in[1];
  const float* value = (const float*)d_in[2];
  const float* boxes = (const float*)d_in[3];
  const float* edge  = (const float*)d_in[4];
  const float* rela  = (const float*)d_in[5];
  const float* Wq  = (const float*)d_in[6];  const float* bq  = (const float*)d_in[7];
  const float* Wk  = (const float*)d_in[8];  const float* bk  = (const float*)d_in[9];
  const float* Wv  = (const float*)d_in[10]; const float* bv  = (const float*)d_in[11];
  const float* Wo  = (const float*)d_in[12]; const float* bo  = (const float*)d_in[13];
  const float* Wrk = (const float*)d_in[14]; const float* brk = (const float*)d_in[15];
  const float* Wrv = (const float*)d_in[16]; const float* brv = (const float*)d_in[17];
  const float* Wst = (const float*)d_in[18]; const float* bst = (const float*)d_in[19];
  const float* Wbx = (const float*)d_in[20]; const float* bbx = (const float*)d_in[21];
  const float* Wvw = (const float*)d_in[22]; const float* bvw = (const float*)d_in[23];
  float* out = (float*)d_out;

  // workspace layout (bf16 tensors), 256B aligned
  char*  ws  = (char*)d_ws;
  size_t off = 0;
  auto alloc = [&](size_t bytes) -> void* {
    void* p = ws + off;
    off += (bytes + 255) & ~(size_t)255;
    return p;
  };
  const size_t QKV_BYTES = (size_t)B_ * H_ * N_ * DK_ * sizeof(bf16_t);  // 2 MiB
  const size_t REL_BYTES = (size_t)B_ * N_ * N_ * DK_ * sizeof(bf16_t);  // 64 MiB
  bf16_t* q_bf = (bf16_t*)alloc(QKV_BYTES);
  bf16_t* k_bf = (bf16_t*)alloc(QKV_BYTES);
  bf16_t* v_bf = (bf16_t*)alloc(QKV_BYTES);
  bf16_t* relk = (bf16_t*)alloc(REL_BYTES);
  bf16_t* relv = (bf16_t*)alloc(REL_BYTES);
  bf16_t* stw  = (bf16_t*)alloc(REL_BYTES);
  bf16_t* bxw  = (bf16_t*)alloc(REL_BYTES);
  bf16_t* x_bf = (bf16_t*)alloc((size_t)B_ * N_ * D_ * sizeof(bf16_t));
  (void)ws_size; (void)in_sizes; (void)n_in; (void)out_size;

  // K1: 3 * 128 * 32 tiles / 8 waves = 1536 blocks
  qkv_proj_kernel<<<1536, 256, 0, stream>>>(query, key, value, Wq, bq, Wk, bk,
                                            Wv, bv, q_bf, k_bf, v_bf);
  // K2: 32768 row stripes / 8 waves = 4096 blocks each
  rel_proj_kernel<<<4096, 256, 0, stream>>>(rela, Wrk, brk, Wrv, brv, relk, relv);
  rel_proj_kernel<<<4096, 256, 0, stream>>>(edge, Wst, bst, Wst, bst, stw, nullptr);
  // K3: 8*256*16 waves / 8 = 4096 blocks
  box_embed_kernel<<<4096, 256, 0, stream>>>(boxes, Wbx, bbx, bxw);
  // K4: (b, i-tile) = 8*16 = 128 blocks, 8 head-waves each
  attn_kernel<<<128, 256, 0, stream>>>(q_bf, k_bf, v_bf, relk, relv, stw, bxw,
                                       Wvw, bvw, x_bf);
  // K5: 128 * 32 tiles / 8 = 512 blocks
  out_proj_kernel<<<512, 256, 0, stream>>>(x_bf, Wo, bo, out);
}
